// GCN_75977971466924
// MI455X (gfx1250) — matrix operand order repaired and checked by reading, not compiled, not run on hardware
//
#include <hip/hip_runtime.h>
#include <hip/hip_bf16.h>

// ---------------------------------------------------------------------------
// GCN forward on MI455X (gfx1250, wave32, WMMA).
//   deg -> deg^-1/2
//   x (f32, K=165) -> xpad (bf16, K=192);  W -> Wt (bf16, transposed, padded)
//   m = A @ W via v_wmma_f32_16x16x32_bf16, weights staged in LDS via
//       global_load_async_to_lds_b128 (ASYNCcnt) once per workgroup
//   agg = self-loop + bias init, then per-edge float4 gather + f32 atomics
//   relu -> bf16 feeds next layer; tiny 64x2 head in f32
// ---------------------------------------------------------------------------

#define N_NODES 100000
#define N_EDGES 1600000
#define IN_CH   165
#define K1PAD   192      // 165 padded to 6 K-steps of 32
#define F1      128
#define K2PAD   128
#define F2      64

typedef __attribute__((ext_vector_type(16))) __bf16 v16bf;
typedef __attribute__((ext_vector_type(8)))  float  v8f;

union FragU {
    struct { uint4 lo, hi; } u;   // 2 x 16B = 32B = 16 bf16
    v16bf v;
};

static __device__ __forceinline__ unsigned short f32_to_bf16(float f) {
    unsigned int u = __float_as_uint(f);
    unsigned int r = 0x7FFFu + ((u >> 16) & 1u);   // round-to-nearest-even
    u += r;
    return (unsigned short)(u >> 16);
}

// ---------------------------------------------------------------------------
// Degree / normalization
// ---------------------------------------------------------------------------
__global__ void k_fill_f32(float* __restrict__ p, float v, int n) {
    int i = blockIdx.x * blockDim.x + threadIdx.x;
    if (i < n) p[i] = v;
}

__global__ void k_deg_count(const int* __restrict__ col, float* __restrict__ deg, int nE) {
    int e = blockIdx.x * blockDim.x + threadIdx.x;
    if (e < nE) atomicAdd(&deg[col[e]], 1.0f);
}

__global__ void k_rsqrt_inplace(float* __restrict__ p, int n) {
    int i = blockIdx.x * blockDim.x + threadIdx.x;
    if (i < n) {
        float d = p[i];
        p[i] = (d > 0.0f) ? __frsqrt_rn(d) : 0.0f;
    }
}

// ---------------------------------------------------------------------------
// bf16 conversion kernels
// ---------------------------------------------------------------------------
__global__ void k_cvt_x(const float* __restrict__ x, unsigned short* __restrict__ xp,
                        int n, int cin, int kpad) {
    long t = (long)blockIdx.x * blockDim.x + threadIdx.x;
    long total = (long)n * kpad;
    if (t >= total) return;
    int node = (int)(t / kpad);
    int k    = (int)(t - (long)node * kpad);
    float v  = (k < cin) ? x[(long)node * cin + k] : 0.0f;
    xp[t] = f32_to_bf16(v);
}

// w [kin][nout] f32 (row-major) -> wt [nout][kpad] bf16 (transposed, zero-padded)
__global__ void k_cvt_w(const float* __restrict__ w, unsigned short* __restrict__ wt,
                        int kin, int nout, int kpad) {
    int t = blockIdx.x * blockDim.x + threadIdx.x;
    int total = nout * kpad;
    if (t >= total) return;
    int nn = t / kpad;
    int k  = t - nn * kpad;
    float v = (k < kin) ? w[(long)k * nout + nn] : 0.0f;
    wt[t] = f32_to_bf16(v);
}

// ---------------------------------------------------------------------------
// WMMA GEMM:  C[M][NT*16] = A[M][KT*32] (bf16) * Bt[NT*16][KT*32]^T (bf16)
// Weights async-DMA'd to LDS once per block (ASYNCcnt); A fragments preloaded
// for the whole K loop; inner loop is pure ds_load_b128 + v_wmma.
// Fragment layouts per CDNA5 ISA 7.12.2.
// ---------------------------------------------------------------------------
template<int NT, int KT>
__global__ __launch_bounds__(256) void k_gemm_wmma(
        const unsigned short* __restrict__ A,
        const unsigned short* __restrict__ Bt,
        float* __restrict__ C,
        int mtiles) {
    constexpr int K  = KT * 32;
    constexpr int Nc = NT * 16;
    constexpr int KL = K + 8;                 // padded LDS row stride (elements)
    __shared__ unsigned short ldsb[Nc * KL];

    // ---- stage Bt into LDS with async DMA (16B chunks, all threads) ------
    constexpr int CHUNKS = (Nc * K) / 8;      // # of 16-byte chunks
    constexpr int KC     = K / 8;             // chunks per row
    for (int c = threadIdx.x; c < CHUNKS; c += 256) {
        int row = c / KC;
        int col = (c - row * KC) * 8;
        unsigned int loff = (unsigned int)(unsigned long long)
                            (const void*)&ldsb[row * KL + col];     // LDS byte addr
        const unsigned short* g = Bt + (size_t)row * K + col;
        asm volatile("global_load_async_to_lds_b128 %0, %1, off"
                     :: "v"(loff), "v"(g) : "memory");
    }

    const int lane  = threadIdx.x & 31;
    const int wave  = threadIdx.x >> 5;
    const int mtile = blockIdx.x * (blockDim.x >> 5) + wave;
    const bool active = (mtile < mtiles);     // wave-uniform
    const int half = lane >> 4;
    const int nrow = lane & 15;               // A m-row / B column / C column

    // ---- preload all A fragments (overlaps the async B staging) ----------
    FragU a[KT];
    if (active) {
        const size_t arow = (size_t)(mtile * 16 + nrow) * K;
#pragma unroll
        for (int kt = 0; kt < KT; ++kt) {
            const unsigned short* ap = A + arow + kt * 32 + half * 8;
            a[kt].u.lo = *(const uint4*)(ap);
            a[kt].u.hi = *(const uint4*)(ap + 16);
        }
    }

    // wait for this wave's async LDS writes, then sync the workgroup
    asm volatile("s_wait_asynccnt 0x0" ::: "memory");
    __syncthreads();
    if (!active) return;

    v8f acc[NT] = {};
#pragma unroll
    for (int kt = 0; kt < KT; ++kt) {
#pragma unroll
        for (int nt = 0; nt < NT; ++nt) {
            // B fragment: column n = lane&15, rows K = 8h..8h+7 and 16+8h..16+8h+7
            const unsigned short* bp = &ldsb[(nt * 16 + nrow) * KL + kt * 32 + 8 * half];
            FragU b;
            b.u.lo = *(const uint4*)(bp);
            b.u.hi = *(const uint4*)(bp + 16);
            acc[nt] = __builtin_amdgcn_wmma_f32_16x16x32_bf16(
                false, a[kt].v, false, b.v, (short)0, acc[nt], false, false);
        }
    }

    // D layout: VGPR r -> row (r + 8*half), col = lane&15
#pragma unroll
    for (int nt = 0; nt < NT; ++nt) {
        float* cp = C + (size_t)(mtile * 16 + half * 8) * Nc + nt * 16 + nrow;
#pragma unroll
        for (int r = 0; r < 8; ++r) {
            cp[(size_t)r * Nc] = acc[nt][r];
        }
    }
}

// ---------------------------------------------------------------------------
// Aggregation: agg[i] = m[i]*dis[i]^2 + b  (self-loop + bias), then
//              agg[col] += m[row]*dis[row]*dis[col]  for every edge (atomics)
// ---------------------------------------------------------------------------
template<int F>
__global__ void k_agg_init(const float* __restrict__ m, const float* __restrict__ dis,
                           const float* __restrict__ b, float* __restrict__ agg, int n) {
    long t = (long)blockIdx.x * blockDim.x + threadIdx.x;
    long total = (long)n * F;
    if (t >= total) return;
    int i = (int)(t >> (F == 128 ? 7 : 6));
    int f = (int)(t & (F - 1));
    float d = dis[i];
    agg[t] = m[t] * d * d + b[f];
}

// one thread per (edge, 4-feature group): float4 gather + 4 f32 atomics
template<int F, int LOGG>   // LOGG = log2(F/4)
__global__ void k_agg_edges(const int* __restrict__ row, const int* __restrict__ col,
                            const float* __restrict__ m, const float* __restrict__ dis,
                            float* __restrict__ agg, int nE) {
    long t = (long)blockIdx.x * blockDim.x + threadIdx.x;
    long total = (long)nE * (F / 4);
    if (t >= total) return;
    int e = (int)(t >> LOGG);
    int g = (int)(t & ((F / 4) - 1));
    int r = row[e];
    int c = col[e];
    float w = dis[r] * dis[c];
    float4 mv = *(const float4*)(m + (long)r * F + g * 4);
    float* ap = agg + (long)c * F + g * 4;
    atomicAdd(ap + 0, mv.x * w);
    atomicAdd(ap + 1, mv.y * w);
    atomicAdd(ap + 2, mv.z * w);
    atomicAdd(ap + 3, mv.w * w);
}

// relu(f32) -> bf16, packed 2 per u32 store
__global__ void k_relu_bf16(const float* __restrict__ a, unsigned int* __restrict__ o, long npairs) {
    long t = (long)blockIdx.x * blockDim.x + threadIdx.x;
    if (t >= npairs) return;
    float2 v = *(const float2*)(a + t * 2);
    float lo = v.x > 0.0f ? v.x : 0.0f;
    float hi = v.y > 0.0f ? v.y : 0.0f;
    o[t] = (unsigned int)f32_to_bf16(lo) | ((unsigned int)f32_to_bf16(hi) << 16);
}

// out[i][c] = sum_f relu(agg2[i][f]) * Wo[f][c] + bo[c],  c in {0,1}
__global__ void k_out_gemm(const float* __restrict__ agg2, const float* __restrict__ Wo,
                           const float* __restrict__ bo, float* __restrict__ out, int n) {
    int i = blockIdx.x * blockDim.x + threadIdx.x;
    if (i >= n) return;
    float a0 = bo[0], a1 = bo[1];
    const float* h = agg2 + (long)i * F2;
#pragma unroll 8
    for (int f = 0; f < F2; ++f) {
        float v = h[f];
        v = v > 0.0f ? v : 0.0f;
        a0 += v * Wo[f * 2 + 0];
        a1 += v * Wo[f * 2 + 1];
    }
    out[(long)i * 2 + 0] = a0;
    out[(long)i * 2 + 1] = a1;
}

// ---------------------------------------------------------------------------
extern "C" void kernel_launch(void* const* d_in, const int* in_sizes, int n_in,
                              void* d_out, int out_size, void* d_ws, size_t ws_size,
                              hipStream_t stream) {
    const float* x   = (const float*)d_in[0];
    const int*   ei  = (const int*)  d_in[1];
    const float* W1  = (const float*)d_in[2];
    const float* b1  = (const float*)d_in[3];
    const float* W2  = (const float*)d_in[4];
    const float* b2  = (const float*)d_in[5];
    const float* Wo  = (const float*)d_in[6];
    const float* bo  = (const float*)d_in[7];
    float* out = (float*)d_out;

    const int* erow = ei;
    const int* ecol = ei + N_EDGES;

    // --- workspace carving (256B aligned), with region reuse --------------
    char* ws = (char*)d_ws;
    size_t off = 0;
    auto carve = [&](size_t bytes) -> char* {
        char* p = ws + off;
        off = (off + bytes + 255) & ~(size_t)255;
        return p;
    };
    float*          dis  = (float*)         carve((size_t)N_NODES * 4);
    unsigned short* xpad = (unsigned short*)carve((size_t)N_NODES * K1PAD * 2); // reused as h1bf
    unsigned short* w1t  = (unsigned short*)carve((size_t)F1 * K1PAD * 2);
    unsigned short* w2t  = (unsigned short*)carve((size_t)F2 * K2PAD * 2);
    float*          m1   = (float*)         carve((size_t)N_NODES * F1 * 4);    // reused as m2
    float*          agg1 = (float*)         carve((size_t)N_NODES * F1 * 4);    // reused as agg2
    unsigned short* h1bf = xpad;           // xpad dead after GEMM1
    float*          m2   = m1;             // m1 dead after layer-1 relu
    float*          agg2 = agg1;           // agg1 dead after layer-1 relu

    const int T = 256;

    // --- deg^-1/2 ---------------------------------------------------------
    k_fill_f32<<<(N_NODES + T - 1) / T, T, 0, stream>>>(dis, 1.0f, N_NODES); // self-loop
    k_deg_count<<<(N_EDGES + T - 1) / T, T, 0, stream>>>(ecol, dis, N_EDGES);
    k_rsqrt_inplace<<<(N_NODES + T - 1) / T, T, 0, stream>>>(dis, N_NODES);

    // --- bf16 conversions -------------------------------------------------
    {
        long tot = (long)N_NODES * K1PAD;
        k_cvt_x<<<(int)((tot + T - 1) / T), T, 0, stream>>>(x, xpad, N_NODES, IN_CH, K1PAD);
    }
    k_cvt_w<<<(F1 * K1PAD + T - 1) / T, T, 0, stream>>>(W1, w1t, IN_CH, F1, K1PAD);
    k_cvt_w<<<(F2 * K2PAD + T - 1) / T, T, 0, stream>>>(W2, w2t, K2PAD, F2, K2PAD);

    // --- layer 1: GEMM (WMMA bf16, LDS-staged weights) + aggregate + relu -
    const int mtiles = N_NODES / 16;                // 6250
    const int gemmBlocks = (mtiles + 7) / 8;        // 8 waves / block
    k_gemm_wmma<8, 6><<<gemmBlocks, T, 0, stream>>>(xpad, w1t, m1, mtiles);

    {
        long tot = (long)N_NODES * F1;
        k_agg_init<F1><<<(int)((tot + T - 1) / T), T, 0, stream>>>(m1, dis, b1, agg1, N_NODES);
        long etot = (long)N_EDGES * (F1 / 4);
        k_agg_edges<F1, 5><<<(int)((etot + T - 1) / T), T, 0, stream>>>(erow, ecol, m1, dis, agg1, N_EDGES);
        k_relu_bf16<<<(int)((tot / 2 + T - 1) / T), T, 0, stream>>>(agg1, (unsigned int*)h1bf, tot / 2);
    }

    // --- layer 2: GEMM (WMMA bf16) + aggregate ----------------------------
    k_gemm_wmma<4, 4><<<gemmBlocks, T, 0, stream>>>(h1bf, w2t, m2, mtiles);
    {
        long tot = (long)N_NODES * F2;
        k_agg_init<F2><<<(int)((tot + T - 1) / T), T, 0, stream>>>(m2, dis, b2, agg2, N_NODES);
        long etot = (long)N_EDGES * (F2 / 4);
        k_agg_edges<F2, 4><<<(int)((etot + T - 1) / T), T, 0, stream>>>(erow, ecol, m2, dis, agg2, N_EDGES);
    }

    // --- output head ------------------------------------------------------
    k_out_gemm<<<(N_NODES + T - 1) / T, T, 0, stream>>>(agg2, Wo, bo, out, N_NODES);
}
